// TPAVIModule_12618613915697
// MI455X (gfx1250) — matrix operand order repaired and checked by reading, not compiled, not run on hardware
//
#include <hip/hip_runtime.h>
#include <hip/hip_bf16.h>
#include <stdint.h>

typedef __attribute__((ext_vector_type(16))) __bf16 v16bf;
typedef __attribute__((ext_vector_type(8)))  float  v8f;

#define N_   8
#define C_   512
#define T_   8
#define HW_  1024
#define S_   8192            // T*H*W tokens per batch
#define CI_  256
#define ZSZ  (N_ * C_ * S_)  // 33,554,432 floats of z output

// ---------------- CDNA5 async global->LDS helpers ----------------
static __device__ __forceinline__ void async_lds_b128(unsigned lds_off, unsigned gl_off,
                                                      const void* base) {
  asm volatile("global_load_async_to_lds_b128 %0, %1, %2"
               :: "v"(lds_off), "v"(gl_off), "s"(base) : "memory");
}
static __device__ __forceinline__ void wait_async_all() {
  asm volatile("s_wait_asynccnt 0" ::: "memory");
}
static __device__ __forceinline__ void wait_async_6() {   // leave next tile (6 issues) in flight
  asm volatile("s_wait_asynccnt 6" ::: "memory");
}
static __device__ __forceinline__ unsigned lds_addr(const void* p) {
  return (unsigned)(unsigned long long)p;  // low 32 bits of flat LDS address = LDS offset
}

// ---------------- k0: zero BN accumulators ----------------
__global__ void k_zero(float* bns) {
  for (int i = threadIdx.x; i < 2 * C_; i += 256) bns[i] = 0.f;
}

// ---------------- k1: xsum[n][t][c] = sum_hw x[n,c,t,:,:] ----------------
__global__ void k_xsum(const float* __restrict__ x, float* __restrict__ XSUM) {
  int row  = blockIdx.x * 8 + (threadIdx.x >> 5);  // row = (n*512 + c)*8 + t
  int lane = threadIdx.x & 31;
  const float* p = x + (size_t)row * HW_;
  float s = 0.f;
  for (int i = lane; i < HW_; i += 32) s += p[i];
  for (int off = 16; off; off >>= 1) s += __shfl_down(s, off, 32);
  if (lane == 0) {
    int t = row & 7;
    int c = (row >> 3) & (C_ - 1);
    int n = row >> 12;
    XSUM[(n * T_ + t) * C_ + c] = s;
  }
}

// ---------------- k2: per-n small chain: audio_temp -> phi, G, M ----------------
__global__ void k_small1(const float* __restrict__ audio, const float* __restrict__ alW,
                         const float* __restrict__ alb, const float* __restrict__ Wg,
                         const float* __restrict__ bg, const float* __restrict__ Wph,
                         const float* __restrict__ bph, const float* __restrict__ XSUM,
                         float* __restrict__ MM, float* __restrict__ out_at) {
  __shared__ float aud[T_ * 128];   // 4KB
  __shared__ float at[T_ * C_];     // 16KB audio_temp for this n
  __shared__ float xs[T_ * C_];     // 16KB xsum for this n
  __shared__ float ph[CI_ * T_];    // 8KB  phi_a
  __shared__ float Gm[T_ * CI_];    // 8KB  G
  const int n = blockIdx.x, tid = threadIdx.x;

  for (int i = tid; i < T_ * 128; i += 256) aud[i] = audio[n * T_ * 128 + i];
  for (int i = tid; i < T_ * C_; i += 256)  xs[i]  = XSUM[n * T_ * C_ + i];
  __syncthreads();

  // audio_temp = relu(audio @ alW^T + alb)   (T x C)
  for (int i = tid; i < T_ * C_; i += 256) {
    int t = i >> 9, c = i & (C_ - 1);
    float v = alb[c];
    for (int k = 0; k < 128; ++k) v += aud[t * 128 + k] * alW[c * 128 + k];
    v = fmaxf(v, 0.f);
    at[i] = v;
    out_at[n * T_ * C_ + i] = v;   // second output of the reference
  }
  __syncthreads();

  // phi[o][t] = Wph . at[t] + bph ;  G[t][o] = Wg . xs[t] + HW*bg
  for (int i = tid; i < CI_ * T_; i += 256) {
    int o = i >> 3, t = i & 7;
    float v = bph[o];
    for (int c = 0; c < C_; ++c) v += Wph[o * C_ + c] * at[t * C_ + c];
    ph[i] = v;
  }
  for (int i = tid; i < T_ * CI_; i += 256) {
    int t = i >> 8, o = i & (CI_ - 1);
    float v = (float)HW_ * bg[o];
    for (int c = 0; c < C_; ++c) v += Wg[o * C_ + c] * xs[t * C_ + c];
    Gm[i] = v;
  }
  __syncthreads();

  // M[k][i] = sum_t phi[k][t] * G[t][i]   (CI x CI)
  for (int idx = tid; idx < CI_ * CI_; idx += 256) {
    int k = idx >> 8, i = idx & (CI_ - 1);
    float v = 0.f;
    for (int t = 0; t < T_; ++t) v += ph[k * T_ + t] * Gm[t * CI_ + i];
    MM[((size_t)n * CI_ + k) * CI_ + i] = v;
  }
}

// ---------------- k3: P = M Wz^T / S ; R = Wth^T P (bf16, transposed) ; r = bth^T P + bz
__global__ void k_small2(const float* __restrict__ MM, const float* __restrict__ Wz,
                         const float* __restrict__ Wth, const float* __restrict__ bth,
                         const float* __restrict__ bz, unsigned short* __restrict__ RT,
                         float* __restrict__ RB) {
  __shared__ float Pl[CI_ * 16];  // 16 output-channel columns
  const int n = blockIdx.x, jb = blockIdx.y * 16, tid = threadIdx.x;
  const float inv = 1.f / (float)S_;

  for (int idx = tid; idx < CI_ * 16; idx += 256) {
    int k = idx & (CI_ - 1), jl = idx >> 8;
    const float* mrow = MM + ((size_t)n * CI_ + k) * CI_;
    const float* wzr  = Wz + (size_t)(jb + jl) * CI_;
    float v = 0.f;
    for (int i = 0; i < CI_; ++i) v += mrow[i] * wzr[i];
    Pl[k * 16 + jl] = v * inv;
  }
  __syncthreads();

  // R[cin][j], stored transposed as RT[n][j][cin] in bf16 (B-matrix for WMMA)
  for (int idx = tid; idx < C_ * 16; idx += 256) {
    int c = idx & (C_ - 1), jl = idx >> 9;
    float v = 0.f;
    for (int k = 0; k < CI_; ++k) v += Wth[k * C_ + c] * Pl[k * 16 + jl];
    RT[((size_t)n * C_ + (jb + jl)) * C_ + c] = __builtin_bit_cast(unsigned short, (__bf16)v);
  }
  if (tid < 16) {
    float v = bz[jb + tid];
    for (int k = 0; k < CI_; ++k) v += bth[k] * Pl[k * 16 + tid];
    RB[n * C_ + jb + tid] = v;
  }
}

// ---------------- k4: heavy WMMA GEMM  W_y[n] = X_n^T R_n + r_n  + BN partial stats
// block: 256 thr (8 waves, wave grid 4x2); tile 128 tokens x 128 cout; K loop 512 by 32
// double-buffered async global->LDS pipeline (6 b128 issues per thread per tile)
__global__ void k_gemm(const float* __restrict__ x, const unsigned short* __restrict__ RT,
                       const float* __restrict__ RB, float* __restrict__ WY,
                       float* __restrict__ BNS, float* __restrict__ BNQ) {
  __shared__ float          As[2][32 * 128];               // A tiles [k][m] fp32, 2 x 16KB
  __shared__ __align__(32) unsigned short Bs[2][128 * 32]; // B tiles [j][k] bf16, 2 x 8KB
  __shared__ float bns[128], bnq[128];

  const int s0 = blockIdx.x * 128;
  const int j0 = blockIdx.y * 128;
  const int n  = blockIdx.z;
  const int tid = threadIdx.x, lane = tid & 31, wave = tid >> 5;
  const int wm = wave >> 1, wn = wave & 1;

  const float* xn          = x + (size_t)n * C_ * S_;
  const unsigned short* rn = RT + (size_t)n * C_ * C_;
  const unsigned AsB[2] = { lds_addr(As[0]), lds_addr(As[1]) };
  const unsigned BsB[2] = { lds_addr(Bs[0]), lds_addr(Bs[1]) };

  // issue one tile (k0) into LDS buffer `buf`: 4 A-slots + 2 B-slots per thread
  auto issue_tile = [&](int k0, int buf) {
#pragma unroll
    for (int i = 0; i < 4; ++i) {
      const int v = tid + i * 256;                     // 1024 slots of 16B: A = 32k x 128m fp32
      const unsigned go = (unsigned)(k0 + (v >> 5)) * (S_ * 4u) +
                          (unsigned)((s0 + (v & 31) * 4) * 4);
      async_lds_b128(AsB[buf] + v * 16u, go, xn);
    }
#pragma unroll
    for (int i = 0; i < 2; ++i) {
      const int v = tid + i * 256;                     // 512 slots of 16B: B = 128j x 32k bf16
      const unsigned go = (unsigned)(((j0 + (v >> 2)) * C_ + k0) * 2 + (v & 3) * 16);
      async_lds_b128(BsB[buf] + v * 16u, go, rn);
    }
  };

  v8f acc[2][4] = {};

  issue_tile(0, 0);
  for (int it = 0; it < 16; ++it) {
    const int cur = it & 1;
    if (it < 15) {
      issue_tile((it + 1) * 32, cur ^ 1);   // prefetch next tile into other buffer
      if (it + 2 < 16)
        __builtin_prefetch(xn + (size_t)(it + 2) * 32 * S_ + s0, 0, 1);
      wait_async_6();                       // current tile landed; next still in flight
    } else {
      wait_async_all();
    }
    __syncthreads();                        // current tile visible to all waves

    // A fragments (16x32 bf16): contiguous k order k = 16*(lane/16) + e
    const float* Ac = As[cur];
    v16bf a[2];
#pragma unroll
    for (int mi = 0; mi < 2; ++mi) {
      const int mcol = wm * 32 + mi * 16 + (lane & 15);
#pragma unroll
      for (int e = 0; e < 16; ++e) {
        const int k = (((lane >> 4) & 1) << 4) | e;
        a[mi][e] = (__bf16)Ac[k * 128 + mcol];
      }
    }
    const int khalf = (lane >> 4) & 1;
    const unsigned short* Bc = Bs[cur];
#pragma unroll
    for (int ni = 0; ni < 4; ++ni) {
      const int jcol = wn * 64 + ni * 16 + (lane & 15);
      v16bf b = *reinterpret_cast<const v16bf*>(&Bc[jcol * 32 + khalf * 16]);
      acc[0][ni] = __builtin_amdgcn_wmma_f32_16x16x32_bf16(false, a[0], false, b,
                                                           (short)0, acc[0][ni], false, false);
      acc[1][ni] = __builtin_amdgcn_wmma_f32_16x16x32_bf16(false, a[1], false, b,
                                                           (short)0, acc[1][ni], false, false);
    }
    __syncthreads();  // all reads of buffer `cur` done before it is refilled at it+1
  }

  if (tid < 128) { bns[tid] = 0.f; bnq[tid] = 0.f; }
  __syncthreads();

  const int rowhalf = ((lane >> 4) & 1) << 3;
#pragma unroll
  for (int ni = 0; ni < 4; ++ni) {
    const int cl = wn * 64 + ni * 16 + (lane & 15);
    const int c  = j0 + cl;
    const float bias = RB[n * C_ + c];
    float s1 = 0.f, s2 = 0.f;
#pragma unroll
    for (int mi = 0; mi < 2; ++mi)
#pragma unroll
      for (int r = 0; r < 8; ++r) {
        const int m = s0 + wm * 32 + mi * 16 + rowhalf + r;
        const float val = acc[mi][ni][r] + bias;
        WY[((size_t)(n * S_ + m)) * C_ + c] = val;
        s1 += val; s2 += val * val;
      }
    atomicAdd(&bns[cl], s1);
    atomicAdd(&bnq[cl], s2);
  }
  __syncthreads();
  if (tid < 128) {
    atomicAdd(&BNS[j0 + tid], bns[tid]);
    atomicAdd(&BNQ[j0 + tid], bnq[tid]);
  }
}

// ---------------- k5: finalize BN scale/shift ----------------
__global__ void k_bnfin(const float* __restrict__ BNS, const float* __restrict__ BNQ,
                        const float* __restrict__ bng, const float* __restrict__ bnb,
                        float* __restrict__ SC, float* __restrict__ SH) {
  const int c = threadIdx.x;
  const float invn = 1.f / (float)(N_ * S_);
  const float mu  = BNS[c] * invn;
  const float var = BNQ[c] * invn - mu * mu;
  const float sc  = bng[c] * rsqrtf(var + 1e-5f);
  SC[c] = sc;
  SH[c] = bnb[c] - mu * sc;
}

// ---------------- k6: fused BN affine + 0.5 blend + LayerNorm(C) + write z ----------------
__global__ void k_fuse(const float* __restrict__ x, const float* __restrict__ WY,
                       const float* __restrict__ SC, const float* __restrict__ SH,
                       const float* __restrict__ lng, const float* __restrict__ lnb,
                       float* __restrict__ out) {
  __shared__ float sc[C_], sh[C_], lg[C_], lb[C_];
  const int tid = threadIdx.x;
  for (int i = tid; i < C_; i += 256) { sc[i] = SC[i]; sh[i] = SH[i]; lg[i] = lng[i]; lb[i] = lnb[i]; }
  __syncthreads();

  const int token = blockIdx.x * 256 + tid;
  const int n = token >> 13, s = token & (S_ - 1);
  const float* wrow = WY + (size_t)token * C_;

  float sum = 0.f, sq = 0.f;
  for (int c = 0; c < C_; ++c) {
    const float xv = x[((size_t)(n * C_ + c)) * S_ + s];
    const float z  = 0.5f * (wrow[c] * sc[c] + sh[c]) + 0.5f * xv;
    sum += z; sq += z * z;
  }
  const float mean = sum * (1.f / C_);
  const float var  = sq * (1.f / C_) - mean * mean;
  const float rstd = rsqrtf(var + 1e-5f);
  for (int c = 0; c < C_; ++c) {
    const float xv = x[((size_t)(n * C_ + c)) * S_ + s];
    const float z  = 0.5f * (wrow[c] * sc[c] + sh[c]) + 0.5f * xv;
    out[((size_t)(n * C_ + c)) * S_ + s] = (z - mean) * rstd * lg[c] + lb[c];
  }
}

// ---------------- host ----------------
extern "C" void kernel_launch(void* const* d_in, const int* in_sizes, int n_in,
                              void* d_out, int out_size, void* d_ws, size_t ws_size,
                              hipStream_t stream) {
  (void)in_sizes; (void)n_in; (void)out_size; (void)ws_size;
  const float* x    = (const float*)d_in[0];
  const float* aud  = (const float*)d_in[1];
  const float* alW  = (const float*)d_in[2];
  const float* alb  = (const float*)d_in[3];
  const float* Wg   = (const float*)d_in[4];
  const float* bg   = (const float*)d_in[5];
  const float* Wth  = (const float*)d_in[6];
  const float* bth  = (const float*)d_in[7];
  const float* Wph  = (const float*)d_in[8];
  const float* bph  = (const float*)d_in[9];
  const float* Wz   = (const float*)d_in[10];
  const float* bz   = (const float*)d_in[11];
  const float* bng  = (const float*)d_in[12];
  const float* bnb  = (const float*)d_in[13];
  const float* lng  = (const float*)d_in[14];
  const float* lnb  = (const float*)d_in[15];

  float* out  = (float*)d_out;            // z (N,C,T,H,W) flat
  float* outA = out + (size_t)ZSZ;        // audio_temp (N,T,C) flat

  // ws layout
  float* WY   = (float*)d_ws;                                   // N*S*C
  float* XSUM = WY + (size_t)N_ * S_ * C_;                      // N*T*C
  float* MM   = XSUM + (size_t)N_ * T_ * C_;                    // N*CI*CI
  unsigned short* RT = (unsigned short*)(MM + (size_t)N_ * CI_ * CI_);  // N*C*C bf16
  float* RB   = (float*)(RT + (size_t)N_ * C_ * C_);            // N*C
  float* BNS  = RB + N_ * C_;                                   // C
  float* BNQ  = BNS + C_;                                       // C
  float* SC   = BNQ + C_;                                       // C
  float* SH   = SC + C_;                                        // C

  k_zero  <<<1, 256, 0, stream>>>(BNS);
  k_xsum  <<<(N_ * C_ * T_) / 8, 256, 0, stream>>>(x, XSUM);
  k_small1<<<N_, 256, 0, stream>>>(aud, alW, alb, Wg, bg, Wph, bph, XSUM, MM, outA);
  k_small2<<<dim3(N_, C_ / 16), 256, 0, stream>>>(MM, Wz, Wth, bth, bz, RT, RB);
  k_gemm  <<<dim3(S_ / 128, C_ / 128, N_), 256, 0, stream>>>(x, RT, RB, WY, BNS, BNQ);
  k_bnfin <<<1, C_, 0, stream>>>(BNS, BNQ, bng, bnb, SC, SH);
  k_fuse  <<<(N_ * S_) / 256, 256, 0, stream>>>(x, WY, SC, SH, lng, lnb, out);
}
